// SDM_47021301957257
// MI455X (gfx1250) — matrix operand order repaired and checked by reading, not compiled, not run on hardware
//
#include <hip/hip_runtime.h>

// ---------------------------------------------------------------------------
// Problem constants (from reference)
// ---------------------------------------------------------------------------
#define BB 1024
#define UU 8
#define SS 50
#define LL 200
#define FF 4
#define EE 32
#define DD 128     // F*E
#define UEUE 256   // U*E

typedef __bf16 bf16_t;
typedef bf16_t v16bf __attribute__((ext_vector_type(16)));
typedef bf16_t v8bf  __attribute__((ext_vector_type(8)));
typedef float  v8f   __attribute__((ext_vector_type(8)));
typedef int    v4i_  __attribute__((ext_vector_type(4)));

__device__ __forceinline__ bf16_t f2bf(float f) {
    unsigned u = __builtin_bit_cast(unsigned, f);
    unsigned r = u + 0x7fffu + ((u >> 16) & 1u);   // round-to-nearest-even
    unsigned short h = (unsigned short)(r >> 16);
    return __builtin_bit_cast(bf16_t, h);
}
__device__ __forceinline__ float sigmoidf_(float x) {
    return 1.0f / (1.0f + __expf(-x));
}

// ---------------------------------------------------------------------------
// CDNA5 async global->LDS copy (ASYNCcnt-tracked), with safe fallback
// ---------------------------------------------------------------------------
#if defined(__has_builtin)
#if __has_builtin(__builtin_amdgcn_global_load_async_to_lds_b128)
#define HAVE_ASYNC_COPY 1
#endif
#endif
#ifndef HAVE_ASYNC_COPY
#define HAVE_ASYNC_COPY 0
#endif

__device__ __forceinline__ void async_copy16(const void* g, void* l) {
#if HAVE_ASYNC_COPY
    // param0: global int4*  (diagnosed type), param1: LDS int4*
    __attribute__((address_space(1))) v4i_* gp =
        (__attribute__((address_space(1))) v4i_*)
        (__attribute__((address_space(1))) void*)(void*)g;
    __attribute__((address_space(3))) v4i_* lp =
        (__attribute__((address_space(3))) v4i_*)
        (__attribute__((address_space(3))) void*)l;
    __builtin_amdgcn_global_load_async_to_lds_b128(gp, lp, 0, 0);
#else
    *(v8bf*)l = *(const v8bf*)g;
#endif
}
__device__ __forceinline__ void wait_copies() {
#if HAVE_ASYNC_COPY
#if __has_builtin(__builtin_amdgcn_s_wait_asynccnt)
    __builtin_amdgcn_s_wait_asynccnt(0);
#else
    asm volatile("s_wait_asynccnt 0" ::: "memory");
#endif
#endif
}

// ---------------------------------------------------------------------------
// Generic WMMA bf16 GEMM:  C(MxN) = A(MxK) @ Bt(NxK)^T [+ bias] [+= C]
// A row-major bf16 (MxK).  Bt row-major bf16 (NxK)  == B column-major.
// C f32 (optionally also bf16 copy). Double-buffered LDS, async copies.
// Tile: 128x64 per workgroup, 8 waves (wave32), each wave 32x32 (2x2 WMMA).
// Requires: M % 128 == 0, N % 64 == 0, K % 32 == 0.
// ---------------------------------------------------------------------------
#define GBM 128
#define GBN 64
#define GBK 32
#define LDS_STR 40   // padded stride in bf16 elems (80 B, 16B-aligned, conflict-free)

__device__ __forceinline__ void stage_tile_A(const bf16_t* __restrict__ A, int lda,
                                             int m0, int k0, bf16_t* sA, int tid) {
    #pragma unroll
    for (int c = 0; c < 2; ++c) {
        int ch  = c * 256 + tid;       // 512 16B chunks (128 rows x 4)
        int row = ch >> 2;
        int cc  = (ch & 3) * 8;
        async_copy16(A + (size_t)(m0 + row) * lda + k0 + cc, &sA[row * LDS_STR + cc]);
    }
}
__device__ __forceinline__ void stage_tile_B(const bf16_t* __restrict__ Bt, int ldbt,
                                             int n0, int k0, bf16_t* sB, int tid) {
    int row = tid >> 2;                // 256 16B chunks (64 rows x 4)
    int cc  = (tid & 3) * 8;
    async_copy16(Bt + (size_t)(n0 + row) * ldbt + k0 + cc, &sB[row * LDS_STR + cc]);
}

__global__ __launch_bounds__(256) void k_gemm_bf16(
    const bf16_t* __restrict__ A, int lda,
    const bf16_t* __restrict__ Bt, int ldbt,
    float* __restrict__ C, int ldc,
    bf16_t* __restrict__ Cbf,           // may be nullptr
    const float* __restrict__ bias,     // may be nullptr
    int M, int N, int K, int accumulate)
{
    __shared__ bf16_t sA[2][GBM * LDS_STR];
    __shared__ bf16_t sB[2][GBN * LDS_STR];

    const int tid  = threadIdx.x;
    const int lane = tid & 31;
    const int wave = tid >> 5;             // 0..7
    const int wm   = (wave & 3) * 32;      // 4 waves along M
    const int wn   = (wave >> 2) * 32;     // 2 waves along N
    const int m0   = blockIdx.x * GBM;
    const int n0   = blockIdx.y * GBN;
    const int lh   = lane >> 4;            // lane half (0/1)
    const int lr   = lane & 15;

    v8f acc[2][2] = {};

    const int ktiles = K / GBK;
    stage_tile_A(A, lda, m0, 0, sA[0], tid);
    stage_tile_B(Bt, ldbt, n0, 0, sB[0], tid);

    for (int kt = 0; kt < ktiles; ++kt) {
        const int cur = kt & 1;
        wait_copies();
        __syncthreads();
        if (kt + 1 < ktiles) {
            stage_tile_A(A, lda, m0, (kt + 1) * GBK, sA[cur ^ 1], tid);
            stage_tile_B(Bt, ldbt, n0, (kt + 1) * GBK, sB[cur ^ 1], tid);
        }

        const bf16_t* cA = sA[cur];
        const bf16_t* cB = sB[cur];

        v16bf afr[2], bfr[2];
        #pragma unroll
        for (int mi = 0; mi < 2; ++mi) {
            // 16-bit A 16x32 layout: lane<16 -> K{0..7,16..23}; lane>=16 -> K{8..15,24..31}
            const bf16_t* base = &cA[(wm + mi * 16 + lr) * LDS_STR];
            v8bf lo = *(const v8bf*)(base + lh * 8);
            v8bf hi = *(const v8bf*)(base + 16 + lh * 8);
            v16bf a;
            #pragma unroll
            for (int e = 0; e < 8; ++e) { a[e] = lo[e]; a[8 + e] = hi[e]; }
            afr[mi] = a;
        }
        #pragma unroll
        for (int ni = 0; ni < 2; ++ni) {
            // 16-bit B 32x16 layout: lane<16 -> K{0..7,16..23}; lane>=16 -> K{8..15,24..31}
            const bf16_t* base = &cB[(wn + ni * 16 + lr) * LDS_STR + lh * 8];
            v8bf lo = *(const v8bf*)(base);
            v8bf hi = *(const v8bf*)(base + 16);
            v16bf b;
            #pragma unroll
            for (int e = 0; e < 8; ++e) { b[e] = lo[e]; b[8 + e] = hi[e]; }
            bfr[ni] = b;
        }
        #pragma unroll
        for (int mi = 0; mi < 2; ++mi)
            #pragma unroll
            for (int ni = 0; ni < 2; ++ni)
                acc[mi][ni] = __builtin_amdgcn_wmma_f32_16x16x32_bf16(
                    false, afr[mi], false, bfr[ni], (short)0, acc[mi][ni],
                    false, false);
    }

    // ---- epilogue: C/D layout: VGPR g, lanes 0-15: M=g,N=lane; lanes 16-31: M=8+g
    #pragma unroll
    for (int mi = 0; mi < 2; ++mi) {
        #pragma unroll
        for (int ni = 0; ni < 2; ++ni) {
            int n = n0 + wn + ni * 16 + lr;
            #pragma unroll
            for (int g = 0; g < 8; ++g) {
                int m = m0 + wm + mi * 16 + lh * 8 + g;
                float v = acc[mi][ni][g];
                if (bias) v += bias[n];
                size_t off = (size_t)m * ldc + n;
                if (accumulate) v += C[off];
                C[off] = v;
                if (Cbf) Cbf[off] = f2bf(v);
            }
        }
    }
}

// ---------------------------------------------------------------------------
// Small helper kernels
// ---------------------------------------------------------------------------
// Cast + transpose: src (KxN, f32, row-major) -> dst rows n: dst[n*dstld + kofs + k]
__global__ void k_cast_t(const float* __restrict__ src, bf16_t* __restrict__ dst,
                         int K, int N, int dstld, int kofs) {
    int g = blockIdx.x * 256 + threadIdx.x;
    if (g < K * N) {
        int k = g / N, n = g % N;
        dst[(size_t)n * dstld + kofs + k] = f2bf(src[g]);
    }
}

// Wt (F,UE,E) -> WtbT ((F*E) x UE): WtbT[i*E+e][r] = Wt[i][r][e]
__global__ void k_build_wtb_t(const float* __restrict__ Wt, bf16_t* __restrict__ dst) {
    int g = blockIdx.x * 256 + threadIdx.x;       // F*UE*E = 32768
    if (g < FF * UEUE * EE) {
        int i = g / (UEUE * EE);
        int r = (g / EE) % UEUE;
        int e = g % EE;
        dst[(size_t)(i * EE + e) * UEUE + r] = f2bf(Wt[g]);
    }
}

__global__ void k_gather_user(const int* __restrict__ prof, const float* __restrict__ emb,
                              float* __restrict__ ue, bf16_t* __restrict__ uebf) {
    int g = blockIdx.x * 256 + threadIdx.x;       // B*UE
    int b = g >> 8, j = g & 255;
    int u = j >> 5, e = j & 31;
    float v = emb[(size_t)prof[b * UU + u] * EE + e];
    ue[g]   = v;
    uebf[g] = f2bf(v);
}

__global__ void k_gather_st(const int* __restrict__ ids, const float* __restrict__ emb,
                            bf16_t* __restrict__ X) {
    size_t g = (size_t)blockIdx.x * 256 + threadIdx.x;   // B*S*D
    int e  = (int)(g & 31);
    int f  = (int)((g >> 5) & 3);
    size_t bs = g >> 7;                                   // b*S + s
    int id = ids[bs * FF + f];
    X[g] = f2bf(emb[(size_t)id * EE + e]);
}

// A_step[b][0:128] = X[b*S+t]; A_step[b][128:256] = h[b]
__global__ void k_pack_lstm(const bf16_t* __restrict__ X, const bf16_t* __restrict__ h,
                            bf16_t* __restrict__ A, int t) {
    int g = blockIdx.x * 256 + threadIdx.x;       // B*256
    int b = g >> 8, j = g & 255;
    A[g] = (j < DD) ? X[((size_t)b * SS + t) * DD + j] : h[b * DD + (j - DD)];
}

__global__ void k_lstm_point(const float* __restrict__ Z, float* __restrict__ c,
                             bf16_t* __restrict__ hbf, bf16_t* __restrict__ Hst, int t) {
    int g = blockIdx.x * 256 + threadIdx.x;       // B*D
    int b = g >> 7, j = g & 127;
    const float* z = Z + (size_t)b * (4 * DD);
    float iv = z[j], fv = z[DD + j], gv = z[2 * DD + j], ov = z[3 * DD + j];
    float cc = sigmoidf_(fv) * c[g] + sigmoidf_(iv) * tanhf(gv);
    float hh = sigmoidf_(ov) * tanhf(cc);
    c[g] = cc;
    bf16_t hb = f2bf(hh);
    hbf[g] = hb;
    Hst[((size_t)b * SS + t) * DD + j] = hb;
}

// per (b, head): 50x50 attention, writes O bf16 (B,S,D)
__global__ __launch_bounds__(64) void k_mha(const float* __restrict__ Q,
                                            const float* __restrict__ Kf,
                                            const float* __restrict__ Vf,
                                            bf16_t* __restrict__ Obf) {
    int b = blockIdx.x >> 2, h = blockIdx.x & 3;
    __shared__ float sK[SS][EE];
    __shared__ float sV[SS][EE];
    for (int idx = threadIdx.x; idx < SS * EE; idx += 64) {
        int r = idx >> 5, e = idx & 31;
        size_t off = ((size_t)b * SS + r) * DD + h * EE + e;
        sK[r][e] = Kf[off];
        sV[r][e] = Vf[off];
    }
    __syncthreads();
    int qi = threadIdx.x;
    if (qi < SS) {
        const float scale = 0.17677669529663687f;  // 1/sqrt(32)
        size_t qoff = ((size_t)b * SS + qi) * DD + h * EE;
        float qreg[EE];
        #pragma unroll
        for (int e = 0; e < EE; ++e) qreg[e] = Q[qoff + e];
        float sc[SS];
        float mx = -3.0e38f;
        for (int kj = 0; kj < SS; ++kj) {
            float s = 0.f;
            #pragma unroll
            for (int e = 0; e < EE; ++e) s += qreg[e] * sK[kj][e];
            s *= scale;
            sc[kj] = s;
            mx = fmaxf(mx, s);
        }
        float sum = 0.f;
        for (int kj = 0; kj < SS; ++kj) { sc[kj] = __expf(sc[kj] - mx); sum += sc[kj]; }
        float inv = 1.f / sum;
        float o[EE];
        #pragma unroll
        for (int e = 0; e < EE; ++e) o[e] = 0.f;
        for (int kj = 0; kj < SS; ++kj) {
            float a = sc[kj] * inv;
            #pragma unroll
            for (int e = 0; e < EE; ++e) o[e] += a * sV[kj][e];
        }
        #pragma unroll
        for (int e = 0; e < EE; ++e) Obf[qoff + e] = f2bf(o[e]);
    }
}

// attention pooling over S: short = softmax(st2 . q) @ st2
__global__ __launch_bounds__(128) void k_pool(const float* __restrict__ st2,
                                              const float* __restrict__ qv,
                                              float* __restrict__ shortf,
                                              bf16_t* __restrict__ shortbf) {
    int b = blockIdx.x;
    __shared__ float sq[DD];
    __shared__ float sw[SS];
    sq[threadIdx.x] = qv[(size_t)b * DD + threadIdx.x];
    __syncthreads();
    int t = threadIdx.x;
    if (t < SS) {
        float s = 0.f;
        const float* row = st2 + ((size_t)b * SS + t) * DD;
        for (int j = 0; j < DD; ++j) s += row[j] * sq[j];
        sw[t] = s;
    }
    __syncthreads();
    if (threadIdx.x == 0) {
        float mx = -3.0e38f;
        for (int i = 0; i < SS; ++i) mx = fmaxf(mx, sw[i]);
        float sum = 0.f;
        for (int i = 0; i < SS; ++i) { sw[i] = __expf(sw[i] - mx); sum += sw[i]; }
        float inv = 1.f / sum;
        for (int i = 0; i < SS; ++i) sw[i] *= inv;
    }
    __syncthreads();
    int e = threadIdx.x;
    float acc = 0.f;
    for (int i = 0; i < SS; ++i) acc += sw[i] * st2[((size_t)b * SS + i) * DD + e];
    shortf[(size_t)b * DD + e]  = acc;
    shortbf[(size_t)b * DD + e] = f2bf(acc);
}

// per (b, field): dedup mask + masked softmax attention over L=200
__global__ __launch_bounds__(256) void k_long(const int* __restrict__ lt,
                                              const float* __restrict__ emb,
                                              const float* __restrict__ UV,
                                              float* __restrict__ longf,
                                              bf16_t* __restrict__ longbf) {
    int b = blockIdx.x >> 2, i = blockIdx.x & 3;
    __shared__ int   sid[LL];
    __shared__ float svec[LL][EE];
    __shared__ float ss[LL];
    __shared__ float suv[EE];
    for (int l = threadIdx.x; l < LL; l += 256)
        sid[l] = lt[((size_t)b * LL + l) * FF + i];
    if (threadIdx.x < EE)
        suv[threadIdx.x] = UV[(size_t)b * DD + i * EE + threadIdx.x];
    __syncthreads();
    for (int idx = threadIdx.x; idx < LL * EE; idx += 256) {
        int l = idx >> 5, e = idx & 31;
        svec[l][e] = emb[(size_t)sid[l] * EE + e];
    }
    __syncthreads();
    int l = threadIdx.x;
    if (l < LL) {
        int myid = sid[l];
        bool keep = true;
        for (int p = 0; p < l; ++p)
            if (sid[p] == myid) { keep = false; break; }
        float s;
        if (keep) {
            s = 0.f;
            #pragma unroll
            for (int e = 0; e < EE; ++e) s += svec[l][e] * suv[e];
        } else {
            s = -1000000000.0f;
        }
        ss[l] = s;
    }
    __syncthreads();
    if (threadIdx.x == 0) {
        float mx = -3.0e38f;
        for (int k = 0; k < LL; ++k) mx = fmaxf(mx, ss[k]);
        float sum = 0.f;
        for (int k = 0; k < LL; ++k) { ss[k] = __expf(ss[k] - mx); sum += ss[k]; }
        float inv = 1.f / sum;
        for (int k = 0; k < LL; ++k) ss[k] *= inv;
    }
    __syncthreads();
    if (threadIdx.x < EE) {
        float acc = 0.f;
        for (int k = 0; k < LL; ++k) acc += ss[k] * svec[k][threadIdx.x];
        size_t off = (size_t)b * DD + i * EE + threadIdx.x;
        longf[off]  = acc;
        longbf[off] = f2bf(acc);
    }
}

__global__ void k_gate(const float* __restrict__ G, const float* __restrict__ shortf,
                       const float* __restrict__ longf, float* __restrict__ out) {
    int g = blockIdx.x * 256 + threadIdx.x;       // B*D
    float gt = sigmoidf_(G[g]);
    out[g] = (1.f - gt) * longf[g] + gt * shortf[g];
}

// ---------------------------------------------------------------------------
// Host launch
// ---------------------------------------------------------------------------
extern "C" void kernel_launch(void* const* d_in, const int* in_sizes, int n_in,
                              void* d_out, int out_size, void* d_ws, size_t ws_size,
                              hipStream_t stream) {
    (void)in_sizes; (void)n_in; (void)out_size; (void)ws_size;

    const int*   user_profile = (const int*)  d_in[0];
    const int*   st_ids       = (const int*)  d_in[1];
    const int*   lt_ids       = (const int*)  d_in[2];
    const float* emb          = (const float*)d_in[3];
    const float* lstm_W       = (const float*)d_in[4];
    const float* lstm_U       = (const float*)d_in[5];
    const float* lstm_b       = (const float*)d_in[6];
    const float* Wq           = (const float*)d_in[7];
    const float* Wk           = (const float*)d_in[8];
    const float* Wv           = (const float*)d_in[9];
    const float* Wo           = (const float*)d_in[10];
    const float* W1           = (const float*)d_in[11];
    const float* b1           = (const float*)d_in[12];
    const float* Wt           = (const float*)d_in[13];
    const float* bt           = (const float*)d_in[14];   // (F,E) flat == D vector
    const float* Wu           = (const float*)d_in[15];
    const float* bu           = (const float*)d_in[16];
    const float* Wsg          = (const float*)d_in[17];
    const float* bsg          = (const float*)d_in[18];
    const float* Wl           = (const float*)d_in[19];
    const float* bl           = (const float*)d_in[20];
    float* out = (float*)d_out;

    // ---- workspace bump allocator
    char* p = (char*)d_ws;
    auto alloc = [&](size_t n) -> char* {
        char* r = p;
        p += (n + 255) & ~(size_t)255;
        return r;
    };
    const size_t BSD = (size_t)BB * SS * DD;

    float*  user_e  = (float*) alloc((size_t)BB * UEUE * 4);
    bf16_t* uebf    = (bf16_t*)alloc((size_t)BB * UEUE * 2);
    bf16_t* WUbT    = (bf16_t*)alloc((size_t)(4 * DD) * UEUE * 2);  // (512 x 256)
    bf16_t* WqbT    = (bf16_t*)alloc((size_t)DD * DD * 2);
    bf16_t* WkbT    = (bf16_t*)alloc((size_t)DD * DD * 2);
    bf16_t* WvbT    = (bf16_t*)alloc((size_t)DD * DD * 2);
    bf16_t* WobT    = (bf16_t*)alloc((size_t)DD * DD * 2);
    bf16_t* W1bT    = (bf16_t*)alloc((size_t)DD * UEUE * 2);
    bf16_t* WtbT    = (bf16_t*)alloc((size_t)DD * UEUE * 2);
    bf16_t* WubT    = (bf16_t*)alloc((size_t)DD * UEUE * 2);
    bf16_t* WsgbT   = (bf16_t*)alloc((size_t)DD * DD * 2);
    bf16_t* WlbT    = (bf16_t*)alloc((size_t)DD * DD * 2);
    bf16_t* Xemb    = (bf16_t*)alloc(BSD * 2);                    // LSTM inputs
    bf16_t* Hst     = (bf16_t*)alloc(BSD * 2);                    // LSTM outputs (all t)
    bf16_t* Astep   = (bf16_t*)alloc((size_t)BB * UEUE * 2);      // [x_t ; h]
    float*  Zbuf    = (float*) alloc((size_t)BB * 4 * DD * 4);
    float*  cbuf    = (float*) alloc((size_t)BB * DD * 4);
    bf16_t* hbf     = (bf16_t*)alloc((size_t)BB * DD * 2);
    float*  Qf      = (float*) alloc(BSD * 4);
    float*  Kf      = (float*) alloc(BSD * 4);
    float*  Vf      = (float*) alloc(BSD * 4);
    bf16_t* Obf     = (bf16_t*)alloc(BSD * 2);
    float*  st2     = (float*) alloc(BSD * 4);                    // MHA output
    float*  qv      = (float*) alloc((size_t)BB * DD * 4);
    float*  shortf  = (float*) alloc((size_t)BB * DD * 4);
    bf16_t* shortbf = (bf16_t*)alloc((size_t)BB * DD * 2);
    float*  UVbuf   = (float*) alloc((size_t)BB * DD * 4);
    float*  longf   = (float*) alloc((size_t)BB * DD * 4);
    bf16_t* longbf  = (bf16_t*)alloc((size_t)BB * DD * 2);
    float*  Gacc    = (float*) alloc((size_t)BB * DD * 4);

    // ---- 1. weight cast-transposes (B operands stored N x K)
    auto castT = [&](const float* s, bf16_t* d, int K, int N, int dstld, int kofs) {
        k_cast_t<<<(K * N + 255) / 256, 256, 0, stream>>>(s, d, K, N, dstld, kofs);
    };
    castT(lstm_W, WUbT, DD, 4 * DD, UEUE, 0);     // rows n of [W;U]^T, k in [0,128)
    castT(lstm_U, WUbT, DD, 4 * DD, UEUE, DD);    // k in [128,256)
    castT(Wq,  WqbT,  DD,   DD, DD,   0);
    castT(Wk,  WkbT,  DD,   DD, DD,   0);
    castT(Wv,  WvbT,  DD,   DD, DD,   0);
    castT(Wo,  WobT,  DD,   DD, DD,   0);
    castT(W1,  W1bT,  UEUE, DD, UEUE, 0);
    castT(Wu,  WubT,  UEUE, DD, UEUE, 0);
    castT(Wsg, WsgbT, DD,   DD, DD,   0);
    castT(Wl,  WlbT,  DD,   DD, DD,   0);
    k_build_wtb_t<<<(FF * UEUE * EE + 255) / 256, 256, 0, stream>>>(Wt, WtbT);

    // ---- 2. gathers
    k_gather_user<<<(BB * UEUE) / 256, 256, 0, stream>>>(user_profile, emb, user_e, uebf);
    k_gather_st<<<(unsigned)(BSD / 256), 256, 0, stream>>>(st_ids, emb, Xemb);

    // ---- 3. LSTM init
    (void)hipMemsetAsync(cbuf, 0, (size_t)BB * DD * 4, stream);
    (void)hipMemsetAsync(hbf,  0, (size_t)BB * DD * 2, stream);

    // ---- 4. LSTM: z = [x_t ; h] @ [W;U] + b ; gates
    dim3 gLstm(BB / GBM, (4 * DD) / GBN);   // 8 x 8
    for (int t = 0; t < SS; ++t) {
        k_pack_lstm<<<(BB * UEUE) / 256, 256, 0, stream>>>(Xemb, hbf, Astep, t);
        k_gemm_bf16<<<gLstm, 256, 0, stream>>>(
            Astep, UEUE, WUbT, UEUE, Zbuf, 4 * DD, nullptr, lstm_b,
            BB, 4 * DD, UEUE, 0);
        k_lstm_point<<<(BB * DD) / 256, 256, 0, stream>>>(Zbuf, cbuf, hbf, Hst, t);
    }

    // ---- 5. MHA
    dim3 gProj((BB * SS) / GBM, DD / GBN);  // 400 x 2
    k_gemm_bf16<<<gProj, 256, 0, stream>>>(Hst, DD, WqbT, DD, Qf, DD, nullptr, nullptr,
                                           BB * SS, DD, DD, 0);
    k_gemm_bf16<<<gProj, 256, 0, stream>>>(Hst, DD, WkbT, DD, Kf, DD, nullptr, nullptr,
                                           BB * SS, DD, DD, 0);
    k_gemm_bf16<<<gProj, 256, 0, stream>>>(Hst, DD, WvbT, DD, Vf, DD, nullptr, nullptr,
                                           BB * SS, DD, DD, 0);
    k_mha<<<BB * FF, 64, 0, stream>>>(Qf, Kf, Vf, Obf);
    k_gemm_bf16<<<gProj, 256, 0, stream>>>(Obf, DD, WobT, DD, st2, DD, nullptr, nullptr,
                                           BB * SS, DD, DD, 0);

    // ---- 6. short-term pooling
    dim3 gSmall(BB / GBM, DD / GBN);        // 8 x 2
    k_gemm_bf16<<<gSmall, 256, 0, stream>>>(uebf, UEUE, W1bT, UEUE, qv, DD, nullptr, b1,
                                            BB, DD, UEUE, 0);
    k_pool<<<BB, 128, 0, stream>>>(st2, qv, shortf, shortbf);

    // ---- 7. long-term fields
    k_gemm_bf16<<<gSmall, 256, 0, stream>>>(uebf, UEUE, WtbT, UEUE, UVbuf, DD, nullptr, bt,
                                            BB, DD, UEUE, 0);
    k_long<<<BB * FF, 256, 0, stream>>>(lt_ids, emb, UVbuf, longf, longbf);

    // ---- 8. gated fusion
    k_gemm_bf16<<<gSmall, 256, 0, stream>>>(uebf, UEUE, WubT, UEUE, Gacc, DD, nullptr, bu,
                                            BB, DD, UEUE, 0);
    k_gemm_bf16<<<gSmall, 256, 0, stream>>>(shortbf, DD, WsgbT, DD, Gacc, DD, nullptr, bsg,
                                            BB, DD, DD, 1);
    k_gemm_bf16<<<gSmall, 256, 0, stream>>>(longbf, DD, WlbT, DD, Gacc, DD, nullptr, bl,
                                            BB, DD, DD, 1);
    k_gate<<<(BB * DD) / 256, 256, 0, stream>>>(Gacc, shortf, longf, out);
}
